// LSTM_43173011259445
// MI455X (gfx1250) — matrix operand order repaired and checked by reading, not compiled, not run on hardware
//
#include <hip/hip_runtime.h>
#include <hip/hip_bf16.h>
#include <math.h>

// ---------------------------------------------------------------------------
// Bidirectional 2-layer LSTM for MI455X (gfx1250), bf16 WMMA + f32 accumulate.
// B=32 S=512 D=512 H=512.
// Round 2: + TDM (tensor_load_to_lds) staging of h in the recurrence,
//          + global_prefetch of next-step gx to hide L2 latency in the
//            serial 512-step chain.
// ---------------------------------------------------------------------------

typedef __bf16 bf16_t;
typedef __attribute__((ext_vector_type(16))) __bf16 v16bf;
typedef __attribute__((ext_vector_type(8)))  __bf16 v8bf;
typedef __attribute__((ext_vector_type(8)))  float  v8f;
typedef __attribute__((ext_vector_type(4)))  unsigned int v4u;
typedef __attribute__((ext_vector_type(4)))  int v4i;
typedef __attribute__((ext_vector_type(8)))  int v8i;

#define B_   32
#define S_   512
#define H_   512
#define D_   512
#define G4H  2048          // 4*H
#define NG   16            // workgroups per direction in recurrence
#define JW   (H_ / NG)     // hidden units owned per WG = 32

#if defined(__has_builtin)
#if __has_builtin(__builtin_amdgcn_tensor_load_to_lds)
#define HAVE_TDM 1
#endif
#endif

static __device__ __forceinline__ v8f wmma_bf16(v16bf a, v16bf b, v8f c) {
  // D = A(16x32) * B(32x16) + C, f32 accumulate
  return __builtin_amdgcn_wmma_f32_16x16x32_bf16(false, a, false, b, (short)0, c,
                                                 false, false);
}

#ifdef HAVE_TDM
// TDM: 1-D tensor load, 16384 x bf16 (32KB) global -> LDS.
// D# per CDNA5 ISA: group0 = {flags, lds_addr,
// global_addr, type=2}; group1 = {data_size=2B, tensor_dim0=tile_dim0=16384}.
static __device__ __forceinline__ void tdm_load_h(const bf16_t* gsrc,
                                                  unsigned lds_off) {
  unsigned long long ga = (unsigned long long)gsrc;
  v4u g0;
  g0[0] = 1u;                                            // count=1 (valid user D#)
  g0[1] = lds_off;                                       // lds_addr [63:32]
  g0[2] = (unsigned)(ga & 0xFFFFFFFFu);                  // global_addr [95:64]
  g0[3] = (unsigned)((ga >> 32) & 0x01FFFFFFu) | (2u << 30); // ga[56:32] | type=2
  v8i g1;
  g1[0] = (int)(1u << 16);          // wg_mask=0, data_size=1 (2 bytes)
  g1[1] = (int)(16384u << 16);      // tensor_dim0[15:0] at bits[63:48]
  g1[2] = (int)(1u << 16);          // tensor_dim0 hi=0 ; tensor_dim1=1
  g1[3] = (int)(16384u << 16);      // tensor_dim1 hi=0 ; tile_dim0=16384
  g1[4] = 0;                        // tile_dim1=0 (unused), tile_dim2=0
  g1[5] = 16384;                    // tensor_dim0_stride low32
  g1[6] = 0;
  g1[7] = 0;
  v4i gz = {0, 0, 0, 0};
#if __clang_major__ >= 23
  v8i gz8 = {0, 0, 0, 0, 0, 0, 0, 0};
  __builtin_amdgcn_tensor_load_to_lds(g0, g1, gz, gz, gz8, 0);
#else
  __builtin_amdgcn_tensor_load_to_lds(g0, g1, gz, gz, 0);
#endif
  __builtin_amdgcn_s_wait_tensorcnt(0);
}
#endif

// ---------------------------------------------------------------------------
// f32 -> bf16 conversion (grid-stride)
// ---------------------------------------------------------------------------
__global__ void cvt_bf16(const float* __restrict__ s, bf16_t* __restrict__ d, int n) {
  int i = blockIdx.x * 256 + threadIdx.x;
  int stride = gridDim.x * 256;
  for (; i < n; i += stride) d[i] = (bf16_t)s[i];
}

// ---------------------------------------------------------------------------
// Zero h-state double buffers (2 layers x 2 dirs x 2 parities) + 4 counters.
// ---------------------------------------------------------------------------
__global__ void init_state(bf16_t* __restrict__ hbufs, unsigned* __restrict__ ctr) {
  int i = blockIdx.x * 256 + threadIdx.x;
  if (i < 8 * B_ * H_) hbufs[i] = (bf16_t)0.0f;
  if (i < 4) ctr[i] = 0u;
}

// ---------------------------------------------------------------------------
// Input projection GEMM: gx[t*B + b][n] = sum_k A[b*S + t][k] * W[n][k] + bias[n]
// A: [B*S, K] bf16 row-major (x or layer-0 output), W: [4H, K] bf16 row-major.
// One wave computes a 16(M) x 64(N) tile; 8 waves / block; grid = 4096 blocks.
// ---------------------------------------------------------------------------
__global__ __launch_bounds__(256)
void xproj(const bf16_t* __restrict__ A, const bf16_t* __restrict__ W,
           const float* __restrict__ bias, float* __restrict__ gx, int K) {
  const int wid  = (blockIdx.x * 256 + threadIdx.x) >> 5;
  const int lane = threadIdx.x & 31;
  const int mt = wid >> 5;       // 0..1023  (M tile of 16 rows)
  const int ng = wid & 31;       // 0..31    (group of 4 N tiles)
  const int n0 = ng * 64;

  v8f acc[4];
#pragma unroll
  for (int u = 0; u < 4; ++u)
#pragma unroll
    for (int r = 0; r < 8; ++r) acc[u][r] = 0.0f;

  const int arow = mt * 16 + (lane & 15);
  const int akb  = (lane >> 4) * 8;    // A K sub-offset: lanes 16-31 hold K+8
  const int bkb  = (lane >> 4) * 8;    // B K sub-offset: lanes 16-31 hold K+8
  const bf16_t* __restrict__ Arow = A + (size_t)arow * K;

  for (int k0 = 0; k0 < K; k0 += 32) {
    v16bf af;
    {
      v8bf lo = *(const v8bf*)(Arow + k0 + akb);
      v8bf hi = *(const v8bf*)(Arow + k0 + akb + 16);
#pragma unroll
      for (int i = 0; i < 8; ++i) { af[i] = lo[i]; af[i + 8] = hi[i]; }
    }
#pragma unroll
    for (int u = 0; u < 4; ++u) {
      const int n = n0 + u * 16 + (lane & 15);
      const bf16_t* wp = W + (size_t)n * K + k0 + bkb;
      v8bf blo = *(const v8bf*)wp;
      v8bf bhi = *(const v8bf*)(wp + 16);
      v16bf bfrag;
#pragma unroll
      for (int i = 0; i < 8; ++i) { bfrag[i] = blo[i]; bfrag[i + 8] = bhi[i]; }
      acc[u] = wmma_bf16(af, bfrag, acc[u]);
    }
  }

  // C/D layout: lane -> n = lane&15 (+tile base), m = (lane>>4)*8 + r
  const int mbase = mt * 16 + (lane >> 4) * 8;
  const int b  = mbase >> 9;      // batch (tiles never straddle a batch row)
  const int tb = mbase & 511;     // time base
#pragma unroll
  for (int u = 0; u < 4; ++u) {
    const int n = n0 + u * 16 + (lane & 15);
    const float bv = bias[n];
#pragma unroll
    for (int r = 0; r < 8; ++r) {
      gx[((size_t)(tb + r) * B_ + b) * G4H + n] = acc[u][r] + bv;
    }
  }
}

// ---------------------------------------------------------------------------
// Recurrence: grid = 32 blocks (16 per direction), 256 threads (8 waves).
// Each WG owns hidden units [j0, j0+32) across all 4 gates (128 gate cols),
// keeps its c-state in LDS, stages full h (32x512 bf16) in LDS each step
// via the Tensor Data Mover. Cross-WG sync: monotonic agent-scope counter
// with s_sleep spin.
// ---------------------------------------------------------------------------
template <int IS_LAST>
__global__ __launch_bounds__(256)
void lstm_recur(const bf16_t* __restrict__ WhhF, const bf16_t* __restrict__ WhhB,
                const float* __restrict__ gxF, const float* __restrict__ gxB,
                bf16_t* __restrict__ hbufs,   // [2 dirs][2 parity][B_*H_]
                bf16_t* __restrict__ out_bf,  // layer0: [B,S,2H] bf16
                float* __restrict__ out_f,    // layer1: [B,S,2H] f32
                float* __restrict__ hn, float* __restrict__ cn,
                unsigned* __restrict__ ctr)   // [2] (per dir)
{
  const int dir = blockIdx.x >> 4;
  const int g   = blockIdx.x & 15;
  const int j0  = g * JW;
  const int tid = threadIdx.x;
  const int wv   = tid >> 5;
  const int lane = tid & 31;
  const int gate  = wv >> 1;                    // 0..3 (i,f,g,o)
  const int jhalf = wv & 1;
  const int ncol  = gate * H_ + j0 + jhalf * 16;  // gate-column base of this wave

  const bf16_t* __restrict__ Whh = dir ? WhhB : WhhF;
  const float*  __restrict__ gx  = dir ? gxB  : gxF;
  bf16_t* hb = hbufs + (size_t)dir * (2 * B_ * H_);
  unsigned* myctr = ctr + dir;

  __shared__ bf16_t h_lds[B_ * H_];        // 32 KB
  __shared__ float  gates[4][B_][JW];      // 16 KB
  __shared__ float  c_lds[B_][JW];         //  4 KB

  for (int i = tid; i < B_ * JW; i += 256) ((float*)c_lds)[i] = 0.0f;

  for (int s = 0; s < S_; ++s) {
    // Wait until all NG workgroups of this direction published h(s).
    if (s > 0 && tid == 0) {
      const unsigned target = (unsigned)(NG * s);
      while (__hip_atomic_load(myctr, __ATOMIC_ACQUIRE, __HIP_MEMORY_SCOPE_AGENT) < target)
        __builtin_amdgcn_s_sleep(1);
    }
    __syncthreads();

    // Stage h(s) into LDS (parity double-buffer in global).
#ifdef HAVE_TDM
    if (wv == 0) {
      tdm_load_h(hb + (size_t)(s & 1) * (B_ * H_),
                 (unsigned)(unsigned long long)(void*)h_lds);
    }
#else
    {
      const v8bf* src = (const v8bf*)(hb + (size_t)(s & 1) * (B_ * H_));
      v8bf* dst = (v8bf*)h_lds;
      for (int i = tid; i < (B_ * H_) / 8; i += 256) dst[i] = src[i];
    }
#endif
    __syncthreads();

    const int t = dir ? (S_ - 1 - s) : s;

    // ---- gates = gx[t] + h @ Whh^T : two 16x16 M-tiles (batch 0-15, 16-31)
    v8f acc0, acc1;
    {
      const int nIdx = ncol + (lane & 15);
      const int mofs = (lane >> 4) * 8;
      const float* gxt = gx + (size_t)t * (B_ * G4H);
#pragma unroll
      for (int r = 0; r < 8; ++r) {
        acc0[r] = gxt[(size_t)(mofs + r) * G4H + nIdx];
        acc1[r] = gxt[(size_t)(16 + mofs + r) * G4H + nIdx];
      }
      // Prefetch next step's gx slice so its L2 latency overlaps the WMMA loop.
      if (s + 1 < S_) {
        const int tn = dir ? (S_ - 2 - s) : (s + 1);
        const float* gxn = gx + (size_t)tn * (B_ * G4H);
        __builtin_prefetch(gxn + (size_t)mofs * G4H + nIdx, 0, 1);
        __builtin_prefetch(gxn + (size_t)(16 + mofs) * G4H + nIdx, 0, 1);
      }
    }
    {
      const int arow = lane & 15;
      const int akb  = (lane >> 4) * 8;
      const int bn   = ncol + (lane & 15);
      const int bkb  = (lane >> 4) * 8;
#pragma unroll 4
      for (int k0 = 0; k0 < H_; k0 += 32) {
        v16bf a0, a1;
        {
          const bf16_t* p0 = h_lds + arow * H_ + k0 + akb;
          const bf16_t* p1 = h_lds + (16 + arow) * H_ + k0 + akb;
          v8bf l0 = *(const v8bf*)p0;
          v8bf u0 = *(const v8bf*)(p0 + 16);
          v8bf l1 = *(const v8bf*)p1;
          v8bf u1 = *(const v8bf*)(p1 + 16);
#pragma unroll
          for (int i = 0; i < 8; ++i) {
            a0[i] = l0[i]; a0[i + 8] = u0[i];
            a1[i] = l1[i]; a1[i + 8] = u1[i];
          }
        }
        const bf16_t* wp = Whh + (size_t)bn * H_ + k0 + bkb;
        v8bf blo = *(const v8bf*)wp;
        v8bf bhi = *(const v8bf*)(wp + 16);
        v16bf bfrag;
#pragma unroll
        for (int i = 0; i < 8; ++i) { bfrag[i] = blo[i]; bfrag[i + 8] = bhi[i]; }
        acc0 = wmma_bf16(a0, bfrag, acc0);
        acc1 = wmma_bf16(a1, bfrag, acc1);
      }
    }

    // Scatter gate tiles to LDS for cross-gate elementwise update.
    {
      const int jloc = jhalf * 16 + (lane & 15);
      const int mofs = (lane >> 4) * 8;
#pragma unroll
      for (int r = 0; r < 8; ++r) {
        gates[gate][mofs + r][jloc]      = acc0[r];
        gates[gate][16 + mofs + r][jloc] = acc1[r];
      }
    }
    __syncthreads();

    // LSTM cell update for 32 batch x 32 hidden units (4 per thread).
    bf16_t* hdst = hb + (size_t)((s + 1) & 1) * (B_ * H_);
#pragma unroll
    for (int k = 0; k < 4; ++k) {
      const int idx = tid + k * 256;
      const int m = idx >> 5;
      const int j = idx & 31;
      const float ig = gates[0][m][j];
      const float fg = gates[1][m][j];
      const float gg = gates[2][m][j];
      const float og = gates[3][m][j];
      const float c_old = c_lds[m][j];
      const float si = 1.0f / (1.0f + __expf(-ig));
      const float sf = 1.0f / (1.0f + __expf(-fg));
      const float so = 1.0f / (1.0f + __expf(-og));
      const float c_new = sf * c_old + si * tanhf(gg);
      const float h_new = so * tanhf(c_new);
      c_lds[m][j] = c_new;
      hdst[m * H_ + j0 + j] = (bf16_t)h_new;
      const size_t oidx = ((size_t)m * S_ + t) * (2 * H_) + dir * H_ + j0 + j;
      if (IS_LAST) {
        out_f[oidx] = h_new;
        if (s == S_ - 1) {
          hn[(size_t)dir * B_ * H_ + m * H_ + j0 + j] = h_new;
          cn[(size_t)dir * B_ * H_ + m * H_ + j0 + j] = c_new;
        }
      } else {
        out_bf[oidx] = (bf16_t)h_new;
      }
    }
    __threadfence();
    __syncthreads();
    if (tid == 0)
      __hip_atomic_fetch_add(myctr, 1u, __ATOMIC_RELEASE, __HIP_MEMORY_SCOPE_AGENT);
  }
}

// ---------------------------------------------------------------------------
// Host-side orchestration (all launches on `stream`, graph-capture safe).
// ---------------------------------------------------------------------------
extern "C" void kernel_launch(void* const* d_in, const int* in_sizes, int n_in,
                              void* d_out, int out_size, void* d_ws, size_t ws_size,
                              hipStream_t stream) {
  (void)in_sizes; (void)n_in; (void)out_size; (void)ws_size;

  const float* x      = (const float*)d_in[0];
  const float* Wih_f0 = (const float*)d_in[1];
  const float* Whh_f0 = (const float*)d_in[2];
  const float* b_f0   = (const float*)d_in[3];
  const float* Wih_b0 = (const float*)d_in[4];
  const float* Whh_b0 = (const float*)d_in[5];
  const float* b_b0   = (const float*)d_in[6];
  const float* Wih_f1 = (const float*)d_in[7];
  const float* Whh_f1 = (const float*)d_in[8];
  const float* b_f1   = (const float*)d_in[9];
  const float* Wih_b1 = (const float*)d_in[10];
  const float* Whh_b1 = (const float*)d_in[11];
  const float* b_b1   = (const float*)d_in[12];

  char* ws = (char*)d_ws;
  size_t off = 0;
  auto take = [&](size_t bytes) -> void* {
    void* p = ws + off;
    off += (bytes + 255) & ~(size_t)255;
    return p;
  };

  const size_t nX   = (size_t)B_ * S_ * D_;       // 8.39M
  const size_t nW0  = (size_t)G4H * 512;          // 1.05M (layer0 Wih/Whh, layer1 Whh)
  const size_t nW1i = (size_t)G4H * 1024;         // 2.10M (layer1 Wih)
  const size_t nGx  = (size_t)S_ * B_ * G4H;      // 33.6M floats
  const size_t nO0  = (size_t)B_ * S_ * 1024;     // 16.8M

  bf16_t* x_bf   = (bf16_t*)take(nX * 2);
  bf16_t* wihf0  = (bf16_t*)take(nW0 * 2);
  bf16_t* whhf0  = (bf16_t*)take(nW0 * 2);
  bf16_t* wihb0  = (bf16_t*)take(nW0 * 2);
  bf16_t* whhb0  = (bf16_t*)take(nW0 * 2);
  bf16_t* wihf1  = (bf16_t*)take(nW1i * 2);
  bf16_t* whhf1  = (bf16_t*)take(nW0 * 2);
  bf16_t* wihb1  = (bf16_t*)take(nW1i * 2);
  bf16_t* whhb1  = (bf16_t*)take(nW0 * 2);
  float*  gxA    = (float*)take(nGx * 4);
  float*  gxB    = (float*)take(nGx * 4);
  bf16_t* out0   = (bf16_t*)take(nO0 * 2);
  bf16_t* hbufs  = (bf16_t*)take((size_t)8 * B_ * H_ * 2);
  unsigned* ctr  = (unsigned*)take(256);

  // init state (h double-buffers + barrier counters)
  init_state<<<dim3(512), dim3(256), 0, stream>>>(hbufs, ctr);

  // f32 -> bf16 conversions
  auto cvt = [&](const float* s, bf16_t* d, size_t n) {
    int blocks = (int)((n + 255) / 256);
    cvt_bf16<<<dim3(blocks), dim3(256), 0, stream>>>(s, d, (int)n);
  };
  cvt(x, x_bf, nX);
  cvt(Wih_f0, wihf0, nW0);  cvt(Whh_f0, whhf0, nW0);
  cvt(Wih_b0, wihb0, nW0);  cvt(Whh_b0, whhb0, nW0);
  cvt(Wih_f1, wihf1, nW1i); cvt(Whh_f1, whhf1, nW0);
  cvt(Wih_b1, wihb1, nW1i); cvt(Whh_b1, whhb1, nW0);

  const dim3 xgrid(4096), xblk(256);

  // ---- Layer 0
  xproj<<<xgrid, xblk, 0, stream>>>(x_bf, wihf0, b_f0, gxA, 512);
  xproj<<<xgrid, xblk, 0, stream>>>(x_bf, wihb0, b_b0, gxB, 512);
  lstm_recur<0><<<dim3(2 * NG), dim3(256), 0, stream>>>(
      whhf0, whhb0, gxA, gxB, hbufs, out0, nullptr, nullptr, nullptr, ctr);

  // ---- Layer 1 (input = out0 bf16, Din = 1024)
  xproj<<<xgrid, xblk, 0, stream>>>(out0, wihf1, b_f1, gxA, 1024);
  xproj<<<xgrid, xblk, 0, stream>>>(out0, wihb1, b_b1, gxB, 1024);

  float* out_f = (float*)d_out;                       // [B,S,2H]
  float* hn    = out_f + (size_t)B_ * S_ * 1024;      // [2,B,H]
  float* cn    = hn + (size_t)2 * B_ * H_;            // [2,B,H]
  lstm_recur<1><<<dim3(2 * NG), dim3(256), 0, stream>>>(
      whhf1, whhb1, gxA, gxB, hbufs + (size_t)4 * B_ * H_, nullptr,
      out_f, hn, cn, ctr + 2);
}
